// Patched_SA_Layer_45097156608034
// MI455X (gfx1250) — matrix operand order repaired and checked by reading, not compiled, not run on hardware
//
#include <hip/hip_runtime.h>
#include <hip/hip_bf16.h>

typedef __attribute__((ext_vector_type(16))) _Float16 v16h;
typedef __attribute__((ext_vector_type(8)))  float    v8f;
typedef __attribute__((ext_vector_type(4)))  _Float16 h4;

#define WIN  128
#define EMBD 128
#define LPITCH 136   // halves per LDS row: 272B -> +4 banks per row, conflict-free frags
#define WMAT (128 * LPITCH)          // halves per weight matrix image
#define WIMG_BYTES (3 * WMAT * 2)    // f16 weight image bytes in workspace

struct __align__(16) Smem {
    _Float16 wq[WMAT];           // Wq row-major [d][e]  (== B^T for Q projection)
    _Float16 wk[WMAT];
    _Float16 wv[WMAT];
    _Float16 xp[WMAT];           // x window f16 [w][e]; re-used as P (softmax probs)
    _Float16 qh[WMAT];           // Q f16 [w][d]
    _Float16 kh[WMAT];           // K f16 [u][d]  (== B^T for scores)
    _Float16 vt[WMAT];           // V^T f16 [d][u] (== B^T for out GEMM)
    float    bqs[128], bks[128], bvs[128];
};

union F16Frag { v16h v; uint4 u[2]; };

// A-matrix 16x32 f16 fragment at (row0.., k0..k0+31), per ISA 7.12.2:
// lane m = row0+(l&15); halves[0..7]=K koff..koff+7, halves[8..15]=K koff+16..+23,
// koff = k0 + 8*(l>>4).
__device__ inline v16h load_a_frag(const _Float16* mat, int row0, int k0, int lane) {
    int m    = row0 + (lane & 15);
    int koff = k0 + ((lane >> 4) << 3);
    const uint4* p = reinterpret_cast<const uint4*>(mat + m * LPITCH + koff);
    F16Frag f;
    f.u[0] = p[0];   // K koff .. koff+7
    f.u[1] = p[2];   // K koff+16 .. koff+23
    return f.v;
}

// B-matrix 32x16 f16 fragment, fed as B^T row-major: lane n = col0+(l&15);
// halves[0..7]=K koff..koff+7, halves[8..15]=K koff+16..+23, koff = k0 + 8*(l>>4).
__device__ inline v16h load_b_frag(const _Float16* bt, int col0, int k0, int lane) {
    int n    = col0 + (lane & 15);
    int koff = k0 + ((lane >> 4) << 3);
    const uint4* p = reinterpret_cast<const uint4*>(bt + n * LPITCH + koff);
    F16Frag f;
    f.u[0] = p[0];
    f.u[1] = p[2];
    return f.v;
}

// ---- Prep: convert f32 weights into an LDS-image-identical f16 block in d_ws ----
__global__ __launch_bounds__(256)
void prep_weights_kernel(const float* __restrict__ Wq, const float* __restrict__ Wk,
                         const float* __restrict__ Wv, _Float16* __restrict__ wsH) {
    int i = blockIdx.x * 256 + threadIdx.x;
    if (i >= 3 * WMAT) return;
    int o = i / WMAT, rem = i % WMAT;
    int r = rem / LPITCH, c = rem % LPITCH;
    const float* W = (o == 0) ? Wq : ((o == 1) ? Wk : Wv);
    _Float16 v = (_Float16)0.f;
    if (c < 128) v = (_Float16)W[r * 128 + c];
    wsH[i] = v;
}

__global__ __launch_bounds__(256)
void win_attn_kernel(const float* __restrict__ x,
                     const float* __restrict__ Wq, const float* __restrict__ bq,
                     const float* __restrict__ Wk, const float* __restrict__ bk,
                     const float* __restrict__ Wv, const float* __restrict__ bv,
                     float* __restrict__ out,
                     const _Float16* __restrict__ wsH, int useWs)
{
    __shared__ Smem s;
    const int tid  = threadIdx.x;
    const int lane = tid & 31;
    const int wave = tid >> 5;
    const int g    = blockIdx.x;                      // window id
    const long base = (long)g * (WIN * EMBD);

    // ---- Phase 0: stage weights (async DMA) + x window (convert) into LDS ----
    if (useWs) {
        // Raw byte copy of the padded f16 weight image: global -> LDS, ASYNCcnt-tracked.
        const int nch = WIMG_BYTES / 16;                          // 16B chunks
        uint32_t ldsBase = (uint32_t)(size_t)(&s.wq[0]);          // low 32 bits = LDS offset
        uint64_t gBase   = (uint64_t)(size_t)wsH;
        for (int c = tid; c < nch; c += 256) {
            uint32_t la = ldsBase + (uint32_t)c * 16u;
            uint64_t ga = gBase + (uint64_t)c * 16u;
            asm volatile("global_load_async_to_lds_b128 %0, %1, off"
                         :: "v"(la), "v"(ga) : "memory");
        }
    } else {
        for (int i = tid; i < 128 * 128; i += 256) {
            int r = i >> 7, c = i & 127;
            s.wq[r * LPITCH + c] = (_Float16)Wq[i];
            s.wk[r * LPITCH + c] = (_Float16)Wk[i];
            s.wv[r * LPITCH + c] = (_Float16)Wv[i];
        }
    }
    // x window: vectorized f32x4 load -> f16x4 LDS store (overlaps the async DMA)
    for (int i4 = tid; i4 < (128 * 128) / 4; i4 += 256) {
        int idx = i4 << 2;
        float4 xv = reinterpret_cast<const float4*>(x + base)[i4];
        int r = idx >> 7, c = idx & 127;
        h4 hh = { (_Float16)xv.x, (_Float16)xv.y, (_Float16)xv.z, (_Float16)xv.w };
        *reinterpret_cast<h4*>(&s.xp[r * LPITCH + c]) = hh;
    }
    if (tid < 128) { s.bqs[tid] = bq[tid]; s.bks[tid] = bk[tid]; s.bvs[tid] = bv[tid]; }
    if (useWs) asm volatile("s_wait_asynccnt 0" ::: "memory");
    __syncthreads();

    const int m0   = wave << 4;          // this wave's 16-row stripe
    const int cn   = lane & 15;          // C/D: column within tile
    const int rsel = (lane >> 4) << 3;   // C/D: +8 row offset for upper half-wave

    // ---- Phase 1: Q = x Wq^T + bq ; K ; V (V stored transposed) ----
    {
        const _Float16* wt[3]  = { s.wq, s.wk, s.wv };
        const float*    bs[3]  = { s.bqs, s.bks, s.bvs };
        _Float16*       dst[3] = { s.qh, s.kh, s.vt };
        #pragma unroll
        for (int o = 0; o < 3; ++o) {
            v8f acc[8];
            #pragma unroll
            for (int t = 0; t < 8; ++t) acc[t] = (v8f){0.f,0.f,0.f,0.f,0.f,0.f,0.f,0.f};
            #pragma unroll
            for (int kk = 0; kk < 128; kk += 32) {
                v16h a = load_a_frag(s.xp, m0, kk, lane);
                #pragma unroll
                for (int t = 0; t < 8; ++t) {
                    v16h b = load_b_frag(wt[o], t << 4, kk, lane);
                    acc[t] = __builtin_amdgcn_wmma_f32_16x16x32_f16(
                        false, a, false, b, (short)0, acc[t], false, false);
                }
            }
            #pragma unroll
            for (int t = 0; t < 8; ++t) {
                int n = (t << 4) + cn;
                float bias = bs[o][n];
                #pragma unroll
                for (int r = 0; r < 8; ++r) {
                    int m = m0 + r + rsel;
                    _Float16 h = (_Float16)(acc[t][r] + bias);
                    if (o == 2) dst[o][n * LPITCH + m] = h;   // V^T[d][u]
                    else        dst[o][m * LPITCH + n] = h;   // Q/K row-major
                }
            }
        }
    }
    __syncthreads();

    // ---- Phase 2: scores = Q K^T, row softmax -> P (f16, aliases xp) ----
    {
        v8f sc[8];
        #pragma unroll
        for (int t = 0; t < 8; ++t) sc[t] = (v8f){0.f,0.f,0.f,0.f,0.f,0.f,0.f,0.f};
        #pragma unroll
        for (int kk = 0; kk < 128; kk += 32) {
            v16h a = load_a_frag(s.qh, m0, kk, lane);
            #pragma unroll
            for (int t = 0; t < 8; ++t) {
                v16h b = load_b_frag(s.kh, t << 4, kk, lane);
                sc[t] = __builtin_amdgcn_wmma_f32_16x16x32_f16(
                    false, a, false, b, (short)0, sc[t], false, false);
            }
        }
        // softmax: row m = m0 + r + rsel lives in one 16-lane half, 8 cols per lane
        #pragma unroll
        for (int r = 0; r < 8; ++r) {
            float mx = sc[0][r];
            #pragma unroll
            for (int t = 1; t < 8; ++t) mx = fmaxf(mx, sc[t][r]);
            #pragma unroll
            for (int d = 1; d < 16; d <<= 1) mx = fmaxf(mx, __shfl_xor(mx, d, 32));
            float p[8], sum = 0.f;
            #pragma unroll
            for (int t = 0; t < 8; ++t) { p[t] = __expf(sc[t][r] - mx); sum += p[t]; }
            #pragma unroll
            for (int d = 1; d < 16; d <<= 1) sum += __shfl_xor(sum, d, 32);
            float inv = 1.f / sum;
            int m = m0 + r + rsel;
            #pragma unroll
            for (int t = 0; t < 8; ++t)
                s.xp[m * LPITCH + (t << 4) + cn] = (_Float16)(p[t] * inv);
        }
    }
    // P rows written/read by the same wave; DS counter ordering suffices (no barrier).

    // ---- Phase 3: out = P V  (B^T = V^T as stored) ----
    {
        v8f oc[8];
        #pragma unroll
        for (int t = 0; t < 8; ++t) oc[t] = (v8f){0.f,0.f,0.f,0.f,0.f,0.f,0.f,0.f};
        #pragma unroll
        for (int kk = 0; kk < 128; kk += 32) {
            v16h a = load_a_frag(s.xp, m0, kk, lane);
            #pragma unroll
            for (int t = 0; t < 8; ++t) {
                v16h b = load_b_frag(s.vt, t << 4, kk, lane);
                oc[t] = __builtin_amdgcn_wmma_f32_16x16x32_f16(
                    false, a, false, b, (short)0, oc[t], false, false);
            }
        }
        float* orow = out + base;
        #pragma unroll
        for (int t = 0; t < 8; ++t) {
            #pragma unroll
            for (int r = 0; r < 8; ++r)
                orow[(m0 + r + rsel) * EMBD + (t << 4) + cn] = oc[t][r];
        }
    }
}

extern "C" void kernel_launch(void* const* d_in, const int* in_sizes, int n_in,
                              void* d_out, int out_size, void* d_ws, size_t ws_size,
                              hipStream_t stream) {
    const float* x  = (const float*)d_in[0];
    const float* Wq = (const float*)d_in[1];
    const float* bq = (const float*)d_in[2];
    const float* Wk = (const float*)d_in[3];
    const float* bk = (const float*)d_in[4];
    const float* Wv = (const float*)d_in[5];
    const float* bv = (const float*)d_in[6];
    float* out = (float*)d_out;

    const int useWs = (ws_size >= (size_t)WIMG_BYTES) ? 1 : 0;
    if (useWs) {
        const int nelem  = 3 * WMAT;
        const int blocks = (nelem + 255) / 256;
        prep_weights_kernel<<<blocks, 256, 0, stream>>>(Wq, Wk, Wv, (_Float16*)d_ws);
    }
    const int nwin = in_sizes[0] / (WIN * EMBD);   // 8*16384*128 / 16384 = 1024
    win_attn_kernel<<<nwin, 256, 0, stream>>>(x, Wq, bq, Wk, bk, Wv, bv, out,
                                              (const _Float16*)d_ws, useWs);
}
